// AttentionLayer_60722247631714
// MI455X (gfx1250) — matrix operand order repaired and checked by reading, not compiled, not run on hardware
//
#include <hip/hip_runtime.h>
#include <hip/hip_bf16.h>

typedef __attribute__((ext_vector_type(16))) __bf16 v16bf;
typedef __attribute__((ext_vector_type(8)))  __bf16 v8bf;
typedef __attribute__((ext_vector_type(8)))  float  v8f;   // also WMMA C/D

// ---- problem constants ----
#define B_   16
#define T_   4096
#define D_   1024
#define NH_  16
#define HD_  64
#define SCALE_ 0.125f           // 1/sqrt(64)

// ---- attention tiling ----
#define CHUNKS     8            // T split into 8 chunks of 512 rows
#define WAVES_     16           // 512-thread workgroup
#define CHUNK_ROWS 512
#define PASS_ROWS  (WAVES_ * 16)   // 256 rows per pass
#define NPART      (CHUNKS * WAVES_)  // 128 partials per (b,h)
#define REC        68           // m, l, ctx[64], pad

// packed f32x8 -> bf16x8 (lowers to v_cvt_pk_bf16_f32 pairs)
static __device__ __forceinline__ v8bf cvt8(v8f x) {
  return __builtin_convertvector(x, v8bf);
}

static __device__ __forceinline__ v8f wmma_bf16(v16bf a, v16bf b, v8f c) {
  // (neg_a, A, neg_b, B, c_mod, C, reuse_a, reuse_b)
  return __builtin_amdgcn_wmma_f32_16x16x32_bf16(false, a, false, b, (short)0, c,
                                                 false, false);
}

static __device__ __forceinline__ v16bf pack16(v8bf lo, v8bf hi) {
  v16bf r;
#pragma unroll
  for (int j = 0; j < 8; ++j) { r[j] = lo[j]; r[8 + j] = hi[j]; }
  return r;
}

// A-fragment (16x32 bf16): lane holds row M=lane&15.
// half-lane hl=lane>>4: elems 0..15 = K base + hl*16 + 0..15.
// rowk must be 32B aligned (true: all K offsets are multiples of 32 floats).
static __device__ __forceinline__ v16bf load_a_frag(const float* __restrict__ rowk,
                                                    int hl) {
  const v8f* p = reinterpret_cast<const v8f*>(rowk + hl * 16);
  v8bf lo = cvt8(p[0]);
  v8bf hi = cvt8(p[1]);   // +8 floats
  return pack16(lo, hi);
}

static __device__ __forceinline__ float silu(float x) {
  return x * __builtin_amdgcn_rcpf(1.0f + __expf(-x));
}

// ---------------------------------------------------------------------------
// Generic tiny-M GEMM: Y[16,Nout] = act( concat(X0[16,K0], X1[16,K-K0]) @ W^T )
// W is [Nout, K] row-major (torch linear weight). One 16-col tile per wave.
// ---------------------------------------------------------------------------
__global__ __launch_bounds__(256) void gemm16_kernel(
    const float* __restrict__ X0, int K0, const float* __restrict__ X1,
    const float* __restrict__ W, float* __restrict__ Y, int K, int Nout, int act) {
  const int tid = threadIdx.x;
  const int wave = tid >> 5, lane = tid & 31;
  const int hl = lane >> 4, l15 = lane & 15;
  const int n = blockIdx.x * 128 + wave * 16 + l15;   // output column this lane owns
  const int K1 = K - K0;

  v8f acc = {};
  for (int kk = 0; kk < K; kk += 32) {
    // A-fragment from concat(X0, X1); 8-runs never straddle K0 (K0 % 32 == 0)
    int ka = kk + hl * 16;
    const float* p0 = (ka < K0) ? (X0 + l15 * K0 + ka) : (X1 + l15 * K1 + (ka - K0));
    int kb = ka + 8;
    const float* p1 = (kb < K0) ? (X0 + l15 * K0 + kb) : (X1 + l15 * K1 + (kb - K0));
    v16bf a = pack16(cvt8(*reinterpret_cast<const v8f*>(p0)),
                     cvt8(*reinterpret_cast<const v8f*>(p1)));

    // B-fragment: lane holds column n, 16 consecutive K split at half-lane.
    const float* wp = W + (size_t)n * K + kk + hl * 16;
    v16bf bb = pack16(cvt8(*reinterpret_cast<const v8f*>(wp)),
                      cvt8(*reinterpret_cast<const v8f*>(wp + 8)));

    acc = wmma_bf16(a, bb, acc);
  }

#pragma unroll
  for (int r = 0; r < 8; ++r) {
    float x = acc[r];
    if (act) x = silu(x);
    Y[(size_t)(r + 8 * hl) * Nout + n] = x; // row = batch (M), col = n
  }
}

// ---------------------------------------------------------------------------
// Fused K/V projection + single-query attention (flash-style partials).
// grid = B*NH*CHUNKS, block = 512 (16 waves). LDS stages the head's weight
// K-panel (64 dims x 64 K, both Wk and Wv) in bf16.
// Partial record per (chunk,wave): [m, l, ctx[64]].
// ---------------------------------------------------------------------------
__global__ __launch_bounds__(512) void attn_kv_kernel(
    const float* __restrict__ enc, const float* __restrict__ Wk,
    const float* __restrict__ Wv, const float* __restrict__ q,
    float* __restrict__ part) {
  __shared__ __bf16 sWk[64][80];   // [n][k-in-panel], padded row (160B stride)
  __shared__ __bf16 sWv[64][80];

  const int tid = threadIdx.x;
  const int wave = tid >> 5, lane = tid & 31;
  const int hl = lane >> 4, l15 = lane & 15;
  const int blk = blockIdx.x;
  const int c = blk % CHUNKS;
  const int h = (blk / CHUNKS) % NH_;
  const int b = blk / (CHUNKS * NH_);

  float qv[4];
#pragma unroll
  for (int t = 0; t < 4; ++t)
    qv[t] = q[b * D_ + h * HD_ + t * 16 + l15] * SCALE_;

  const float* wkbase = Wk + (size_t)(h * HD_) * D_;
  const float* wvbase = Wv + (size_t)(h * HD_) * D_;

  float m_run = -3.0e38f, l_run = 0.0f;
  float cacc[4] = {0.f, 0.f, 0.f, 0.f};

  for (int pass = 0; pass < CHUNK_ROWS / PASS_ROWS; ++pass) {
    const int row = c * CHUNK_ROWS + pass * PASS_ROWS + wave * 16 + l15;
    const float* arow = enc + ((size_t)b * T_ + row) * D_;

    v8f accK[4], accV[4];
#pragma unroll
    for (int t = 0; t < 4; ++t) { accK[t] = (v8f){}; accV[t] = (v8f){}; }

    for (int kk = 0; kk < D_; kk += 64) {
      // prefetch the enc stream ahead (global_prefetch_b8 path)
      if (kk + 128 < D_) __builtin_prefetch(arow + kk + 128, 0, 3);

      __syncthreads();
      {   // cooperative panel load: 512 threads, 8 K-floats each per matrix
        const int n = tid >> 3;
        const int k8 = (tid & 7) * 8;
        const v8f* pk = reinterpret_cast<const v8f*>(wkbase + (size_t)n * D_ + kk + k8);
        const v8f* pv = reinterpret_cast<const v8f*>(wvbase + (size_t)n * D_ + kk + k8);
        *reinterpret_cast<v8bf*>(&sWk[n][k8]) = cvt8(*pk);
        *reinterpret_cast<v8bf*>(&sWv[n][k8]) = cvt8(*pv);
      }
      __syncthreads();

#pragma unroll
      for (int ks = 0; ks < 64; ks += 32) {
        v16bf a = load_a_frag(arow + kk + ks, hl);
#pragma unroll
        for (int nt = 0; nt < 4; ++nt) {
          const v16bf* bk =
              reinterpret_cast<const v16bf*>(&sWk[nt * 16 + l15][ks + hl * 16]);
          const v16bf* bv =
              reinterpret_cast<const v16bf*>(&sWv[nt * 16 + l15][ks + hl * 16]);
          accK[nt] = wmma_bf16(a, *bk, accK[nt]);
          accV[nt] = wmma_bf16(a, *bv, accV[nt]);
        }
      }
    }

    // scores: s[M] = sum_N k[M,N] * q[N]; C-layout: VGPR r, lane -> M=r+8*hl, N=l15
    float sp[8];
#pragma unroll
    for (int r = 0; r < 8; ++r) {
      float s = accK[0][r] * qv[0] + accK[1][r] * qv[1] +
                accK[2][r] * qv[2] + accK[3][r] * qv[3];
      s += __shfl_xor(s, 1, 32);
      s += __shfl_xor(s, 2, 32);
      s += __shfl_xor(s, 4, 32);
      s += __shfl_xor(s, 8, 32);
      sp[r] = s;   // score for row r + 8*hl, replicated in half-lane group
    }
    float mt = sp[0];
#pragma unroll
    for (int r = 1; r < 8; ++r) mt = fmaxf(mt, sp[r]);
    mt = fmaxf(mt, __shfl_xor(mt, 16, 32));

    const float m_new = fmaxf(m_run, mt);
    float p[8], lt = 0.f;
#pragma unroll
    for (int r = 0; r < 8; ++r) { p[r] = __expf(sp[r] - m_new); lt += p[r]; }
    lt += __shfl_xor(lt, 16, 32);

    const float scale = __expf(m_run - m_new);
#pragma unroll
    for (int t = 0; t < 4; ++t) {
      float ct = 0.f;
#pragma unroll
      for (int r = 0; r < 8; ++r) ct += p[r] * accV[t][r];
      ct += __shfl_xor(ct, 16, 32);       // add other half's 8 rows
      cacc[t] = cacc[t] * scale + ct;      // ctx for N = t*16 + l15
    }
    l_run = l_run * scale + lt;
    m_run = m_new;
  }

  float* rec =
      part + (size_t)((b * NH_ + h) * NPART + (c * WAVES_ + wave)) * REC;
  if (lane == 0) { rec[0] = m_run; rec[1] = l_run; }
  if (lane < 16) {
#pragma unroll
    for (int t = 0; t < 4; ++t) rec[2 + t * 16 + lane] = cacc[t];
  }
}

// ---------------------------------------------------------------------------
// Combine NPART partials per (b,h) -> ctx[b, h*64 + n]
// ---------------------------------------------------------------------------
__global__ __launch_bounds__(128) void attn_reduce_kernel(
    const float* __restrict__ part, float* __restrict__ ctx) {
  __shared__ float red[NPART];
  __shared__ float wgt[NPART];
  const int tid = threadIdx.x;
  const int h = blockIdx.x % NH_;
  const int b = blockIdx.x / NH_;
  const float* base = part + (size_t)((b * NH_ + h) * NPART) * REC;

  const float mi = base[(size_t)tid * REC + 0];
  const float li = base[(size_t)tid * REC + 1];

  red[tid] = mi;
  __syncthreads();
  for (int s = NPART / 2; s > 0; s >>= 1) {
    if (tid < s) red[tid] = fmaxf(red[tid], red[tid + s]);
    __syncthreads();
  }
  const float mstar = red[0];
  __syncthreads();

  const float w = __expf(mi - mstar);
  wgt[tid] = w;
  red[tid] = li * w;
  __syncthreads();
  for (int s = NPART / 2; s > 0; s >>= 1) {
    if (tid < s) red[tid] += red[tid + s];
    __syncthreads();
  }
  const float invL = __builtin_amdgcn_rcpf(red[0]);
  __syncthreads();

  if (tid < HD_) {
    float s = 0.f;
    for (int i = 0; i < NPART; ++i) s += base[(size_t)i * REC + 2 + tid] * wgt[i];
    ctx[b * D_ + h * HD_ + tid] = s * invL;
  }
}

// ---------------------------------------------------------------------------
extern "C" void kernel_launch(void* const* d_in, const int* in_sizes, int n_in,
                              void* d_out, int out_size, void* d_ws,
                              size_t ws_size, hipStream_t stream) {
  (void)in_sizes; (void)n_in; (void)out_size; (void)ws_size;
  const float* dec_h = (const float*)d_in[0];
  const float* enc   = (const float*)d_in[1];
  const float* Wq    = (const float*)d_in[2];
  const float* Wk    = (const float*)d_in[3];
  const float* Wv    = (const float*)d_in[4];
  const float* W1    = (const float*)d_in[5];
  const float* W2    = (const float*)d_in[6];

  float* ws   = (float*)d_ws;
  float* qbuf = ws;                 // 16*1024
  float* ctx  = ws + 16384;         // 16*1024
  float* hbuf = ws + 32768;         // 16*4096
  float* part = ws + 98304;         // 16*16*128*68 floats (~8.5 MB)

  // q = dec_h @ Wq^T
  gemm16_kernel<<<D_ / 128, 256, 0, stream>>>(dec_h, D_, nullptr, Wq, qbuf,
                                              D_, D_, 0);
  // fused K/V projection + attention partials
  attn_kv_kernel<<<B_ * NH_ * CHUNKS, 512, 0, stream>>>(enc, Wk, Wv, qbuf, part);
  // combine partials -> ctx (head-concatenated)
  attn_reduce_kernel<<<B_ * NH_, 128, 0, stream>>>(part, ctx);
  // h = silu(concat(dec_h, ctx) @ W1^T)
  gemm16_kernel<<<(4 * D_) / 128, 256, 0, stream>>>(dec_h, D_, ctx, W1, hbuf,
                                                    2 * D_, 4 * D_, 1);
  // out = h @ W2^T
  gemm16_kernel<<<D_ / 128, 256, 0, stream>>>(hbuf, 4 * D_, nullptr, W2,
                                              (float*)d_out, 4 * D_, D_, 0);
}
